// MultiHeadSelfAttention_80573586473711
// MI455X (gfx1250) — compile-verified, matrix operand order checked by reading, not run on hardware
//
#include <hip/hip_runtime.h>
#include <hip/hip_bf16.h>

// ---------------------------------------------------------------------------
// Multi-head causal self-attention for gfx1250 (MI455X), bf16 WMMA pipeline
// with TDM (tensor_load_to_lds) staging of GEMM weight tiles.
// B=4, S=2048, D=1024, H=16, dk=64.
// ---------------------------------------------------------------------------

#define BB   4
#define SS   2048
#define DD   1024
#define HH   16
#define DK   64
#define MM   (BB * SS)          // 8192 rows

typedef __attribute__((ext_vector_type(16))) __bf16       v16bf;
typedef __attribute__((ext_vector_type(8)))  __bf16       v8bf;
typedef __attribute__((ext_vector_type(8)))  float        v8f;
typedef __attribute__((ext_vector_type(4)))  unsigned int u32x4;
typedef __attribute__((ext_vector_type(4)))  int          i32x4;
typedef __attribute__((ext_vector_type(8)))  int          i32x8;

#if __has_builtin(__builtin_amdgcn_tensor_load_to_lds) && \
    __has_builtin(__builtin_amdgcn_s_wait_tensorcnt)
#define USE_TDM 1
#else
#define USE_TDM 0
#endif

// Build a 16x32 bf16 WMMA A/B fragment from two contiguous 16-byte chunks.
// Per ISA layout: elements 0..7 cover K = h*8..h*8+7, elements 8..15 cover
// K = 16+h*8..16+h*8+7 (h = lane/16).
__device__ __forceinline__ v16bf load_frag(const __bf16* lo, const __bf16* hi) {
    v8bf a = *(const v8bf*)lo;
    v8bf b = *(const v8bf*)hi;
    v16bf r;
#pragma unroll
    for (int i = 0; i < 8; ++i) { r[i] = a[i]; r[i + 8] = b[i]; }
    return r;
}

#if USE_TDM
// Issue one TDM 2D tile load: tile_d0 x tile_d1 elements (bf16) from a row-major
// [tensor_d1, tensor_d0] tensor (row stride = row_stride elems) into LDS at
// lds_addr, packed contiguously (tile_d0 elems per row).
__device__ __forceinline__ void tdm_load_tile(unsigned lds_addr, const void* gaddr,
                                              int tensor_d0, int tensor_d1,
                                              int tile_d0, int tile_d1,
                                              int row_stride) {
    unsigned long long ga = (unsigned long long)(uintptr_t)gaddr;
    u32x4 g0;
    g0[0] = 1u;                                   // count=1, user descriptor
    g0[1] = lds_addr;                             // LDS byte address
    g0[2] = (unsigned)(ga & 0xFFFFFFFFu);         // global_addr[31:0]
    g0[3] = (unsigned)((ga >> 32) & 0x1FFFFFFu) | (2u << 30);  // [56:32] | type=2
    i32x8 g1;
    g1[0] = (1 << 16);                            // data_size=1 (2 bytes), wg_mask=0
    g1[1] = (tensor_d0 & 0xFFFF) << 16;           // tensor_dim0[15:0]
    g1[2] = ((tensor_d0 >> 16) & 0xFFFF) | ((tensor_d1 & 0xFFFF) << 16);
    g1[3] = ((tensor_d1 >> 16) & 0xFFFF) | ((tile_d0 & 0xFFFF) << 16);
    g1[4] = (tile_d1 & 0xFFFF);                   // tile_dim1 (tile_dim2 = 0)
    g1[5] = row_stride;                           // tensor_dim0_stride[31:0]
    g1[6] = 0;                                    // stride hi / dim1_stride lo
    g1[7] = 0;
    i32x4 z4 = {0, 0, 0, 0};
#if __has_include(<hip/amd_detail/amd_gfx1250_TDM.h>)
    i32x8 z8 = {0, 0, 0, 0, 0, 0, 0, 0};
    __builtin_amdgcn_tensor_load_to_lds(g0, g1, z4, z4, z8, 0);
#else
    __builtin_amdgcn_tensor_load_to_lds(g0, g1, z4, z4, 0);
#endif
}
#endif

// ---------------------------------------------------------------------------
// Elementwise f32 -> bf16 cast
// ---------------------------------------------------------------------------
__global__ void cast_bf16_kernel(const float* __restrict__ src,
                                 __bf16* __restrict__ dst, int n) {
    int i = blockIdx.x * blockDim.x + threadIdx.x;
    if (i < n) dst[i] = (__bf16)src[i];
}

// ---------------------------------------------------------------------------
// Tiled bf16 GEMM:  C[m,n] = sum_k A[m,k] * W[n,k]
// A: [M,K] row-major bf16, W: [N,K] row-major bf16, C: [M,N] f32.
// Block = 128 threads (4 waves), covers 64 rows x 64 cols.
// The 64x32 W tile (shared by all 4 waves) is staged in LDS, double-buffered,
// via the Tensor Data Mover; A fragments are register double-buffered.
// ---------------------------------------------------------------------------
__global__ void gemm_bf16_kernel(const __bf16* __restrict__ A,
                                 const __bf16* __restrict__ W,
                                 float* __restrict__ C,
                                 int M, int N, int K) {
    __shared__ alignas(16) __bf16 wtile[2][64 * 32];   // [buf][n-in-tile][k]

    const int lane = threadIdx.x & 31;
    const int wave = threadIdx.x >> 5;
    const int nl = lane & 15;
    const int hl = lane >> 4;

    const int rbase = blockIdx.x * 64 + wave * 16;
    const int cbase = blockIdx.y * 64;

    v8f acc[4] = {};

    const __bf16* arow = A + (size_t)(rbase + nl) * K;
    const __bf16* wtileg = W + (size_t)cbase * K;       // tile origin in W

#if USE_TDM
    const unsigned lds0 = (unsigned)(uintptr_t)&wtile[0][0];
    const unsigned lds1 = (unsigned)(uintptr_t)&wtile[1][0];
    if (wave == 0) tdm_load_tile(lds0, wtileg, K, N, 32, 64, K);
#endif

    v16bf afc = load_frag(arow + hl * 8, arow + 16 + hl * 8);

    for (int kk = 0; kk < K; kk += 32) {
        const int cur = (kk >> 5) & 1;

#if USE_TDM
        if (wave == 0) __builtin_amdgcn_s_wait_tensorcnt(0);
        __syncthreads();                       // W tile `cur` ready for all waves
        if (kk + 32 < K && wave == 0)
            tdm_load_tile(cur ? lds0 : lds1, wtileg + (kk + 32), K, N, 32, 64, K);
#endif

        // Register double-buffer the next A fragment (overlaps with WMMAs).
        v16bf afn = afc;
        if (kk + 32 < K) {
            __builtin_prefetch(arow + kk + 96, 0, 3);
            afn = load_frag(arow + kk + 32 + hl * 8, arow + kk + 48 + hl * 8);
        }

#pragma unroll
        for (int c = 0; c < 4; ++c) {
#if USE_TDM
            const __bf16* brow = &wtile[cur][(c * 16 + nl) * 32];
#else
            const __bf16* brow = W + (size_t)(cbase + c * 16 + nl) * K + kk;
#endif
            v16bf bf_ = load_frag(brow + hl * 8, brow + 16 + hl * 8);
            acc[c] = __builtin_amdgcn_wmma_f32_16x16x32_bf16(
                false, afc, false, bf_, (short)0, acc[c], false, false);
        }

        afc = afn;
#if USE_TDM
        __syncthreads();                       // all waves done reading `cur`
#endif
    }

#pragma unroll
    for (int c = 0; c < 4; ++c) {
#pragma unroll
        for (int r = 0; r < 8; ++r) {
            C[(size_t)(rbase + r + 8 * hl) * N + cbase + c * 16 + nl] = acc[c][r];
        }
    }
}

// ---------------------------------------------------------------------------
// RoPE + relayout: src f32 [B,S,H*dk]  ->  dst bf16 [B,H,S,dk]
// ---------------------------------------------------------------------------
__global__ void rope_relayout_kernel(const float* __restrict__ src,
                                     const int* __restrict__ pos,
                                     __bf16* __restrict__ dst) {
    int t = blockIdx.x * blockDim.x + threadIdx.x;
    const int TOT = BB * HH * SS * (DK / 2);
    if (t >= TOT) return;
    int j = t & 31;  int t2 = t >> 5;
    int s = t2 % SS; t2 /= SS;
    int h = t2 % HH; int b = t2 / HH;

    float p = (float)pos[s];
    float inv = __expf(-(float)j * (9.210340371976184f / 32.0f));  // theta^{-j/32}
    float ang = p * inv;
    float sn, cs;
    __sincosf(ang, &sn, &cs);

    const float* in = src + (size_t)(b * SS + s) * DD + h * DK + 2 * j;
    float x1 = in[0];
    float x2 = in[1];

    __bf16* o = dst + ((size_t)(b * HH + h) * SS + s) * DK + 2 * j;
    o[0] = (__bf16)(x1 * cs - x2 * sn);
    o[1] = (__bf16)(x1 * sn + x2 * cs);
}

// ---------------------------------------------------------------------------
// V relayout: src f32 [B,S,H*dk] -> dst bf16 [B,H,dk,S]
// ---------------------------------------------------------------------------
__global__ void v_relayout_kernel(const float* __restrict__ src,
                                  __bf16* __restrict__ dst) {
    int t = blockIdx.x * blockDim.x + threadIdx.x;
    const int TOT = BB * HH * DK * SS;
    if (t >= TOT) return;
    int s = t % SS;  int t2 = t / SS;
    int d = t2 % DK; t2 /= DK;
    int h = t2 % HH; int b = t2 / HH;
    dst[((size_t)(b * HH + h) * DK + d) * SS + s] =
        (__bf16)src[(size_t)(b * SS + s) * DD + h * DK + d];
}

// ---------------------------------------------------------------------------
// Flash-attention style causal attention.
// q,k: bf16 [B,H,S,dk]; vt: bf16 [B,H,dk,S]; out (concat): bf16 [B,S,H*dk].
// Block = 128 threads (4 waves); wave w handles 16 query rows.
// K and V fragment loads are issued together at the top of the key-block loop
// so they overlap the score WMMAs and the softmax VALU work.
// ---------------------------------------------------------------------------
__global__ void attention_kernel(const __bf16* __restrict__ q,
                                 const __bf16* __restrict__ k,
                                 const __bf16* __restrict__ vt,
                                 __bf16* __restrict__ concat) {
    __shared__ alignas(16) __bf16 pbuf[4][16 * 32];

    const int lane = threadIdx.x & 31;
    const int wave = threadIdx.x >> 5;
    const int nl = lane & 15;
    const int hl = lane >> 4;

    const int qbase = blockIdx.x * 64 + wave * 16;
    const int head  = blockIdx.y;
    const int b     = blockIdx.z;
    const int bh    = b * HH + head;
    const float scale = 0.125f;   // 1/sqrt(64)

    __bf16* pb = &pbuf[wave][0];

    const __bf16* qrow = q + ((size_t)bh * SS + qbase + nl) * DK;
    v16bf qa0 = load_frag(qrow + hl * 8,      qrow + 16 + hl * 8);
    v16bf qa1 = load_frag(qrow + 32 + hl * 8, qrow + 48 + hl * 8);

    float mstate[8], lstate[8];
#pragma unroll
    for (int r = 0; r < 8; ++r) { mstate[r] = -1.0e30f; lstate[r] = 0.0f; }
    v8f o[4] = {};

    for (int kb = 0; kb < qbase + 16; kb += 32) {
        // ---- issue all global loads for this block up front -----------------
        const __bf16* krow0 = k + ((size_t)bh * SS + kb + nl) * DK;
        const __bf16* krow1 = krow0 + (size_t)16 * DK;
        v16bf kf0a = load_frag(krow0 + hl * 8,      krow0 + 16 + hl * 8);
        v16bf kf0b = load_frag(krow0 + 32 + hl * 8, krow0 + 48 + hl * 8);
        v16bf kf1a = load_frag(krow1 + hl * 8,      krow1 + 16 + hl * 8);
        v16bf kf1b = load_frag(krow1 + 32 + hl * 8, krow1 + 48 + hl * 8);

        v16bf vf[4];
#pragma unroll
        for (int c = 0; c < 4; ++c) {
            const __bf16* vrow = vt + ((size_t)bh * DK + c * 16 + nl) * SS + kb;
            vf[c] = load_frag(vrow + hl * 8, vrow + 16 + hl * 8);
        }

        // ---- scores: S[16q x 32k] via 4 WMMAs -------------------------------
        v8f s0 = {}, s1 = {};
        s0 = __builtin_amdgcn_wmma_f32_16x16x32_bf16(false, qa0, false, kf0a, (short)0, s0, false, false);
        s0 = __builtin_amdgcn_wmma_f32_16x16x32_bf16(false, qa1, false, kf0b, (short)0, s0, false, false);
        s1 = __builtin_amdgcn_wmma_f32_16x16x32_bf16(false, qa0, false, kf1a, (short)0, s1, false, false);
        s1 = __builtin_amdgcn_wmma_f32_16x16x32_bf16(false, qa1, false, kf1b, (short)0, s1, false, false);

        // ---- online softmax -------------------------------------------------
        float corr[8];
#pragma unroll
        for (int r = 0; r < 8; ++r) {
            const int grow = qbase + r + 8 * hl;
            const int col0 = kb + nl;
            const int col1 = kb + 16 + nl;
            float v0 = s0[r] * scale;
            float v1 = s1[r] * scale;
            if (col0 > grow) v0 = -1.0e9f;
            if (col1 > grow) v1 = -1.0e9f;

            float mx = fmaxf(v0, v1);
#pragma unroll
            for (int off = 1; off < 16; off <<= 1)
                mx = fmaxf(mx, __shfl_xor(mx, off, 32));
            float mnew = fmaxf(mstate[r], mx);
            float c = __expf(mstate[r] - mnew);
            float p0 = __expf(v0 - mnew);
            float p1 = __expf(v1 - mnew);
            float ps = p0 + p1;
#pragma unroll
            for (int off = 1; off < 16; off <<= 1)
                ps += __shfl_xor(ps, off, 32);
            lstate[r] = lstate[r] * c + ps;
            mstate[r] = mnew;
            corr[r] = c;

            pb[(r + 8 * hl) * 32 + nl]      = (__bf16)p0;
            pb[(r + 8 * hl) * 32 + 16 + nl] = (__bf16)p1;
        }

#pragma unroll
        for (int c = 0; c < 4; ++c)
#pragma unroll
            for (int r = 0; r < 8; ++r) o[c][r] *= corr[r];

        // wave-local LDS RAW: wait for DS stores, read P as A-layout fragment
        asm volatile("s_wait_dscnt 0x0" ::: "memory");
        const __bf16* prow = pb + nl * 32;
        v16bf pa = load_frag(prow + hl * 8, prow + 16 + hl * 8);

        // ---- O += P @ V -----------------------------------------------------
#pragma unroll
        for (int c = 0; c < 4; ++c) {
            o[c] = __builtin_amdgcn_wmma_f32_16x16x32_bf16(
                false, pa, false, vf[c], (short)0, o[c], false, false);
        }
    }

    // ---- normalize + write to concat [B,S,H*dk] (bf16) ----------------------
#pragma unroll
    for (int r = 0; r < 8; ++r) {
        const float inv = 1.0f / lstate[r];
        const int grow = qbase + r + 8 * hl;
#pragma unroll
        for (int c = 0; c < 4; ++c) {
            concat[(size_t)(b * SS + grow) * DD + head * DK + c * 16 + nl] =
                (__bf16)(o[c][r] * inv);
        }
    }
}

// ---------------------------------------------------------------------------
// Host launcher
// ---------------------------------------------------------------------------
extern "C" void kernel_launch(void* const* d_in, const int* in_sizes, int n_in,
                              void* d_out, int out_size, void* d_ws, size_t ws_size,
                              hipStream_t stream) {
    (void)in_sizes; (void)n_in; (void)out_size; (void)ws_size;

    const float* x    = (const float*)d_in[0];
    const int*   tpos = (const int*)d_in[1];
    const float* wq   = (const float*)d_in[2];
    const float* wk   = (const float*)d_in[3];
    const float* wv   = (const float*)d_in[4];
    const float* wo   = (const float*)d_in[5];
    float* out        = (float*)d_out;

    char* ws = (char*)d_ws;
    size_t off = 0;
    auto alloc = [&](size_t bytes) { char* p = ws + off; off += (bytes + 255) & ~(size_t)255; return p; };

    __bf16* xb   = (__bf16*)alloc((size_t)MM * DD * 2);
    __bf16* wqb  = (__bf16*)alloc((size_t)DD * DD * 2);
    __bf16* wkb  = (__bf16*)alloc((size_t)DD * DD * 2);
    __bf16* wvb  = (__bf16*)alloc((size_t)DD * DD * 2);
    __bf16* wob  = (__bf16*)alloc((size_t)DD * DD * 2);
    float*  tmpf = (float*) alloc((size_t)MM * DD * 4);
    __bf16* qb   = (__bf16*)alloc((size_t)MM * DD * 2);
    __bf16* kb   = (__bf16*)alloc((size_t)MM * DD * 2);
    __bf16* vtb  = (__bf16*)alloc((size_t)MM * DD * 2);
    __bf16* cb   = (__bf16*)alloc((size_t)MM * DD * 2);

    const int nX = MM * DD;
    const int nW = DD * DD;

    cast_bf16_kernel<<<(nX + 255) / 256, 256, 0, stream>>>(x,  xb,  nX);
    cast_bf16_kernel<<<(nW + 255) / 256, 256, 0, stream>>>(wq, wqb, nW);
    cast_bf16_kernel<<<(nW + 255) / 256, 256, 0, stream>>>(wk, wkb, nW);
    cast_bf16_kernel<<<(nW + 255) / 256, 256, 0, stream>>>(wv, wvb, nW);
    cast_bf16_kernel<<<(nW + 255) / 256, 256, 0, stream>>>(wo, wob, nW);

    dim3 ggrid(MM / 64, DD / 64);   // (128, 16)
    dim3 gblk(128);

    const int nRope = BB * HH * SS * (DK / 2);
    const int nVT   = BB * HH * DK * SS;

    gemm_bf16_kernel<<<ggrid, gblk, 0, stream>>>(xb, wqb, tmpf, MM, DD, DD);
    rope_relayout_kernel<<<(nRope + 255) / 256, 256, 0, stream>>>(tmpf, tpos, qb);

    gemm_bf16_kernel<<<ggrid, gblk, 0, stream>>>(xb, wkb, tmpf, MM, DD, DD);
    rope_relayout_kernel<<<(nRope + 255) / 256, 256, 0, stream>>>(tmpf, tpos, kb);

    gemm_bf16_kernel<<<ggrid, gblk, 0, stream>>>(xb, wvb, tmpf, MM, DD, DD);
    v_relayout_kernel<<<(nVT + 255) / 256, 256, 0, stream>>>(tmpf, vtb);

    dim3 agrid(SS / 64, HH, BB);   // (32, 16, 4)
    attention_kernel<<<agrid, gblk, 0, stream>>>(qb, kb, vtb, cb);

    gemm_bf16_kernel<<<ggrid, gblk, 0, stream>>>(cb, wob, out, MM, DD, DD);
}
